// Net_79139067396690
// MI455X (gfx1250) — matrix operand order repaired and checked by reading, not compiled, not run on hardware
//
#include <hip/hip_runtime.h>
#include <hip/hip_bf16.h>
#include <math.h>

typedef __attribute__((ext_vector_type(16))) _Float16 v16h;
typedef __attribute__((ext_vector_type(8)))  _Float16 v8h;
typedef __attribute__((ext_vector_type(8)))  float    v8f;

#define D_IN      784
#define KSTEPS1   25          // ceil(784/32) -> K padded to 800
#define HID       256
#define KSTEPS2   8           // 256/32
#define NT1       16          // 256/16 column tiles layer 1
#define C_OUT     10
#define P_KEEP    0.5f
#define INV_SQRT_Z 0.031622776601683794f  // 1/sqrt(1000)

// fragment record: per (ntile,kstep) 32 lanes x 16 halfs (32B/lane), lane-major
#define W1_FRAG_V16 (NT1 * KSTEPS1 * 32)          // 12800 v16h records
#define W2_FRAG_V16 (KSTEPS2 * 32)                // 256 v16h records

// ---- deterministic cheap normal draw: popcount(binomial(32,1/2)) ----------
// mean 16, var 8 -> (popc - 16)/sqrt(8). Noise term is scaled by 1/sqrt(1000),
// so a 33-level discrete normal is statistically adequate and ~10 VALU ops.
__device__ __forceinline__ float hash_normal(unsigned row, unsigned col, unsigned salt) {
    unsigned s = row * 0x9E3779B9u ^ (col + 0x7F4A7C15u) * 0x85EBCA6Bu ^ salt * 0xC2B2AE35u;
    s ^= s >> 16; s *= 0x7FEB352Du; s ^= s >> 15; s *= 0x846CA68Bu; s ^= s >> 16;
    return ((float)__popc(s) - 16.0f) * 0.35355339059327376f;
}

// B-fragment element mapping for v_wmma_f32_16x16x32_f16 (32x16 B):
// lane l: n = l&15 ; element e -> K = (l>=16 ? 8 : 0) + e + (e>=8 ? 8 : 0)
__device__ __forceinline__ int bfrag_k(int lane, int elem) {
    return ((lane >> 4) << 3) + elem + ((elem >> 3) << 3);
}

// ---------------- pack W1 into fragment order (f16); W1^2 derived in-regs ---
__global__ __launch_bounds__(256) void pack_w1(const float* __restrict__ W1,
                                               _Float16* __restrict__ w1f) {
    int tid  = blockIdx.x * 256 + threadIdx.x;          // 16*25*32 = 12800 records
    if (tid >= W1_FRAG_V16) return;
    int lane = tid & 31;
    int rec  = tid >> 5;
    int ks   = rec % KSTEPS1;
    int nt   = rec / KSTEPS1;
    int n    = lane & 15;
    size_t base = (size_t)tid * 16;
    #pragma unroll
    for (int e = 0; e < 16; ++e) {
        int k = ks * 32 + bfrag_k(lane, e);
        float w = (k < D_IN) ? W1[(size_t)(nt * 16 + n) * D_IN + k] : 0.0f;
        w1f[base + e] = (_Float16)w;
    }
}

// ---------------- pack W2 (N padded 10->16) ---------------------------------
__global__ __launch_bounds__(256) void pack_w2(const float* __restrict__ W2,
                                               _Float16* __restrict__ w2f) {
    int tid  = threadIdx.x;                              // 8*32 = 256 records
    int lane = tid & 31;
    int ks   = tid >> 5;
    int n    = lane & 15;
    size_t base = (size_t)tid * 16;
    #pragma unroll
    for (int e = 0; e < 16; ++e) {
        int k = ks * 32 + bfrag_k(lane, e);
        float w = (n < C_OUT) ? W2[(size_t)n * HID + k] : 0.0f;
        w2f[base + e] = (_Float16)w;
    }
}

// ---------------- fused 2-layer moment-propagation MLP ----------------------
#define XPAD 8    // row padding: 40 halfs = 80B stride (16B-aligned, conflict-reducing)
#define HPAD 8    // 264 halfs = 528B stride

__global__ __launch_bounds__(256) void fused_mlp(const float* __restrict__ x,
                                                 const v16h* __restrict__ w1f,
                                                 const v16h* __restrict__ w2f,
                                                 float* __restrict__ out) {
    __shared__ __align__(16) _Float16 xtile[2][16][32 + XPAD];   // ping-pong
    __shared__ __align__(16) _Float16 htile[16][HID + HPAD];

    const int tid  = threadIdx.x;
    const int lane = tid & 31;
    const int wave = tid >> 5;
    const int row0 = blockIdx.x * 16;

    // A-fragment addressing (16-bit A 16x32): lane l -> m = l&15,
    // vgprs 0-3: K = (l>=16?8:0)+0..7 ; vgprs 4-7: K = 16+(l>=16?8:0)+0..7
    const int am    = lane & 15;
    const int abase = (lane >> 4) << 3;

    // x staging map: thread t handles float2 at (row = t>>4, col = (t&15)*2)
    const int lr = tid >> 4;
    const int lc = (tid & 15) << 1;
    const float* xrow = x + (size_t)(row0 + lr) * D_IN;

    const int nt0 = wave * 2;
    const int nt1 = wave * 2 + 1;

    // B-fragment streaming pointers (stride: 32 records = 1KB per K-step)
    const v16h* pb0 = w1f + (size_t)nt0 * KSTEPS1 * 32 + lane;
    const v16h* pb1 = w1f + (size_t)nt1 * KSTEPS1 * 32 + lane;

    v8f accM0 = {}; v8f accM1 = {}; v8f accV0 = {}; v8f accV1 = {};

    // ---- prologue: stage tile 0 ----
    {
        const float2 p = *(const float2*)(xrow + lc);    // ks=0: always in range
        xtile[0][lr][lc]     = (_Float16)p.x;
        xtile[0][lr][lc + 1] = (_Float16)p.y;
    }
    __syncthreads();

    for (int ks = 0; ks < KSTEPS1; ++ks) {
        // ---- fetch tile ks+1 into registers (overlaps with WMMA below) ----
        float n0 = 0.0f, n1 = 0.0f;
        const int kn = (ks + 1) * 32 + lc;
        if (ks + 1 < KSTEPS1 && kn < D_IN) {             // pair-aligned: both valid
            const float2 p = *(const float2*)(xrow + kn);
            n0 = p.x; n1 = p.y;
        }
        if (ks + 2 < KSTEPS1 && kn + 32 < D_IN)
            __builtin_prefetch(xrow + kn + 32, 0, 3);

        // ---- A fragment from LDS buffer ks&1 (2x ds_load_b128) ----
        v8h alo = *(const v8h*)&xtile[ks & 1][am][abase];
        v8h ahi = *(const v8h*)&xtile[ks & 1][am][16 + abase];
        v16h a;
        #pragma unroll
        for (int i = 0; i < 8; ++i) { a[i] = alo[i]; a[8 + i] = ahi[i]; }
        v16h a2 = a * a;                                  // x^2 fragment, in-regs

        // ---- B fragments (packed, L2-resident); squares derived in-regs ----
        v16h b0  = *pb0;  pb0 += 32;
        v16h b1  = *pb1;  pb1 += 32;
        v16h bs0 = b0 * b0;
        v16h bs1 = b1 * b1;

        accM0 = __builtin_amdgcn_wmma_f32_16x16x32_f16(false, a,  false, b0,  (short)0, accM0, false, false);
        accM1 = __builtin_amdgcn_wmma_f32_16x16x32_f16(false, a,  false, b1,  (short)0, accM1, false, false);
        accV0 = __builtin_amdgcn_wmma_f32_16x16x32_f16(false, a2, false, bs0, (short)0, accV0, false, false);
        accV1 = __builtin_amdgcn_wmma_f32_16x16x32_f16(false, a2, false, bs1, (short)0, accV1, false, false);

        // ---- store tile ks+1 into the other buffer; single barrier ----
        if (ks + 1 < KSTEPS1) {
            xtile[(ks + 1) & 1][lr][lc]     = (_Float16)n0;
            xtile[(ks + 1) & 1][lr][lc + 1] = (_Float16)n1;
        }
        __syncthreads();
    }

    // ---- layer-1 epilogue: h = relu(P*mean + sqrt(P(1-P)*var)*eps/sqrt(Z)) ----
    // C/D layout: vgpr r, lane l -> m = r + (l>=16 ? 8 : 0), n = l&15
    {
        const int moff = (lane >> 4) << 3;
        const int n    = lane & 15;
        #pragma unroll
        for (int r = 0; r < 8; ++r) {
            int m = r + moff;
            unsigned grow = row0 + m;
            {
                unsigned gcol = nt0 * 16 + n;
                float hv = P_KEEP * accM0[r]
                         + 0.5f * sqrtf(fmaxf(accV0[r], 0.0f))
                           * hash_normal(grow, gcol, 1u) * INV_SQRT_Z;
                htile[m][gcol] = (_Float16)fmaxf(hv, 0.0f);
            }
            {
                unsigned gcol = nt1 * 16 + n;
                float hv = P_KEEP * accM1[r]
                         + 0.5f * sqrtf(fmaxf(accV1[r], 0.0f))
                           * hash_normal(grow, gcol, 1u) * INV_SQRT_Z;
                htile[m][gcol] = (_Float16)fmaxf(hv, 0.0f);
            }
        }
    }
    __syncthreads();

    // ---- layer 2 on wave 0: out[16,10] = moments(h @ W2^T), no relu ----
    if (wave == 0) {
        v8f oM = {}; v8f oV = {};
        const v16h* pb = w2f + lane;
        #pragma unroll 1
        for (int ks = 0; ks < KSTEPS2; ++ks) {
            const int cb = ks * 32;
            v8h alo = *(const v8h*)&htile[am][cb + abase];
            v8h ahi = *(const v8h*)&htile[am][cb + 16 + abase];
            v16h a;
            #pragma unroll
            for (int i = 0; i < 8; ++i) { a[i] = alo[i]; a[8 + i] = ahi[i]; }
            v16h a2 = a * a;                              // h^2 fragment
            v16h b  = *pb;  pb += 32;
            v16h bs = b * b;                              // W2^2 fragment
            oM = __builtin_amdgcn_wmma_f32_16x16x32_f16(false, a,  false, b,  (short)0, oM, false, false);
            oV = __builtin_amdgcn_wmma_f32_16x16x32_f16(false, a2, false, bs, (short)0, oV, false, false);
        }
        const int moff = (lane >> 4) << 3;
        const int n    = lane & 15;
        #pragma unroll
        for (int r = 0; r < 8; ++r) {
            int m = r + moff;
            unsigned grow = row0 + m;
            float val = P_KEEP * oM[r]
                      + 0.5f * sqrtf(fmaxf(oV[r], 0.0f))
                        * hash_normal(grow, (unsigned)n, 2u) * INV_SQRT_Z;
            if (n < C_OUT)
                out[(size_t)grow * C_OUT + n] = val;
        }
    }
}

extern "C" void kernel_launch(void* const* d_in, const int* in_sizes, int n_in,
                              void* d_out, int out_size, void* d_ws, size_t ws_size,
                              hipStream_t stream) {
    const float* x  = (const float*)d_in[0];
    const float* W1 = (const float*)d_in[1];
    const float* W2 = (const float*)d_in[2];
    float* out = (float*)d_out;

    const int B = in_sizes[0] / D_IN;   // 65536

    _Float16* w1f = (_Float16*)d_ws;
    _Float16* w2f = w1f + (size_t)W1_FRAG_V16 * 16;

    // pack weights into WMMA B-fragment order (cheap; L2-resident afterwards)
    pack_w1<<<(W1_FRAG_V16 + 255) / 256, 256, 0, stream>>>(W1, w1f);
    pack_w2<<<1, 256, 0, stream>>>(W2, w2f);

    fused_mlp<<<B / 16, 256, 0, stream>>>(x,
                                          (const v16h*)w1f,
                                          (const v16h*)w2f,
                                          out);
}
